// SpatialTail_2843268350069
// MI455X (gfx1250) — matrix operand order repaired and checked by reading, not compiled, not run on hardware
//
#include <hip/hip_runtime.h>

// Problem constants (from reference): B=4, C=256, H=W=64, N=H*W=4096
#define Bb 4
#define Cc 256
#define Nn 4096
#define CK 256   // K-chunk for the fused softmax+GEMM kernel
#define MT 64    // attention rows per block in fused kernel
#define APAD 8   // LDS row padding (halves) -> 4-bank skew per row

typedef __bf16 bf16;
typedef __attribute__((ext_vector_type(16))) __bf16 v16bf;
typedef __attribute__((ext_vector_type(8)))  __bf16 v8bf;
typedef __attribute__((ext_vector_type(8)))  float  v8f;

__device__ inline v16bf join16(v8bf lo, v8bf hi) {
  return __builtin_shufflevector(lo, hi, 0,1,2,3,4,5,6,7,8,9,10,11,12,13,14,15);
}

// ---------------------------------------------------------------------------
// K1: transpose + convert  x[b][c][n] (f32)  ->  x_t[b][n][c] (bf16)
// ---------------------------------------------------------------------------
__global__ __launch_bounds__(256)
void xpose_kernel(const float* __restrict__ x, bf16* __restrict__ xt) {
  __shared__ float tile[16][17];
  int b = blockIdx.z, c0 = blockIdx.y * 16, n0 = blockIdx.x * 16;
  int tx = threadIdx.x, ty = threadIdx.y;
  tile[ty][tx] = x[((size_t)(b * Cc + c0 + ty)) * Nn + n0 + tx];
  __syncthreads();
  xt[((size_t)(b * Nn + n0 + ty)) * Cc + c0 + tx] = (bf16)tile[tx][ty];
}

// ---------------------------------------------------------------------------
// K2: value_t[b][c][n] (bf16) = sum_k W[c][k]*x[b][k][n] + bias[c]
//     A = W (f32 -> bf16 on the fly), B = x_t rows (K-contiguous per lane)
// ---------------------------------------------------------------------------
__global__ __launch_bounds__(256)
void value_kernel(const bf16* __restrict__ xt, const float* __restrict__ W,
                  const float* __restrict__ bias, bf16* __restrict__ vt) {
  int b  = blockIdx.y;
  int n0 = blockIdx.x * 16;
  int wv = threadIdx.x >> 5, lane = threadIdx.x & 31;
  int lm = lane & 15, hi = lane >> 4;

  const bf16* xrow = xt + ((size_t)(b * Nn + n0 + lm)) * Cc;

  for (int t = 0; t < 2; ++t) {
    int ct   = wv * 2 + t;            // 8 waves x 2 = 16 channel tiles
    int crow = ct * 16 + lm;          // A-matrix row (lanes 16-31 mirror 0-15)
    const float* wrow = W + (size_t)crow * Cc;

    v8f acc;
    #pragma unroll
    for (int i = 0; i < 8; ++i) acc[i] = 0.0f;

    for (int k0 = 0; k0 < Cc; k0 += 32) {
      // B tile: 16 contiguous halves, lanes>=16 take K+16
      v16bf bt = *(const v16bf*)(xrow + k0 + hi * 16);
      // A tile: halves[0..15]=K=k0+hi*16..
      const float* p0 = wrow + k0 + hi * 16;
      const float* p1 = wrow + k0 + 8 + hi * 16;
      v16bf at;
      #pragma unroll
      for (int i = 0; i < 8; ++i) { at[i] = (bf16)p0[i]; at[8 + i] = (bf16)p1[i]; }
      acc = __builtin_amdgcn_wmma_f32_16x16x32_bf16(false, at, false, bt,
                                                    (short)0, acc, false, false);
    }
    #pragma unroll
    for (int r = 0; r < 8; ++r) {
      int c = ct * 16 + r + hi * 8;   // D: M = r + 8*(lane>=16)
      float v = acc[r] + bias[c];
      vt[((size_t)(b * Cc + c)) * Nn + n0 + lm] = (bf16)v;
    }
  }
}

// ---------------------------------------------------------------------------
// K3: fused softmax(energy) @ value^T, epilogue gamma*out + 2x
//     Block = 512 threads (16 wave32), 64 attention rows of one batch.
//     Attention never touches HBM: bf16 exp-chunks live in LDS only.
// ---------------------------------------------------------------------------
__global__ __launch_bounds__(512)
void fused_kernel(const float* __restrict__ energy, const bf16* __restrict__ vt,
                  const float* __restrict__ x, const float* __restrict__ gamma,
                  float* __restrict__ out) {
  __shared__ alignas(32) bf16 attn[MT][CK + APAD];   // 64 x 264 halves = 33 KB
  __shared__ float rowmax[MT];
  __shared__ float rowrcp[MT];

  int b   = blockIdx.y;
  int m0  = blockIdx.x * MT;
  int tid = threadIdx.x;
  int wv  = tid >> 5, lane = tid & 31;
  int lm  = lane & 15, hi = lane >> 4;

  const float* ebase = energy + (size_t)b * Nn * Nn;

  // -------- Phase 1: per-row max and exp-sum (4 rows per wave) --------
  for (int rr = 0; rr < 4; ++rr) {
    int row = wv * 4 + rr;
    const float* er = ebase + (size_t)(m0 + row) * Nn;
    float mx = -3.4e38f;
    for (int i = 0; i < Nn / 128; ++i) {
      float4 v = *(const float4*)(er + i * 128 + lane * 4);
      mx = fmaxf(mx, fmaxf(fmaxf(v.x, v.y), fmaxf(v.z, v.w)));
    }
    for (int o = 16; o > 0; o >>= 1) mx = fmaxf(mx, __shfl_xor(mx, o, 32));
    float s = 0.0f;
    for (int i = 0; i < Nn / 128; ++i) {     // re-read hits L2
      float4 v = *(const float4*)(er + i * 128 + lane * 4);
      s += __expf(v.x - mx) + __expf(v.y - mx) + __expf(v.z - mx) + __expf(v.w - mx);
    }
    for (int o = 16; o > 0; o >>= 1) s += __shfl_xor(s, o, 32);
    if (lane == 0) { rowmax[row] = mx; rowrcp[row] = 1.0f / s; }
  }
  __syncthreads();

  v8f acc[4];
  #pragma unroll
  for (int rt = 0; rt < 4; ++rt)
    #pragma unroll
    for (int i = 0; i < 8; ++i) acc[rt][i] = 0.0f;

  int ct = wv;                                   // 16 waves -> 16 channel tiles
  const bf16* vbase = vt + ((size_t)(b * Cc + ct * 16 + lm)) * Nn + hi * 16;

  int arow = tid >> 3;                           // 0..63 (row for LDS fill)
  int asub = tid & 7;                            // 8 threads per row

  // -------- Phase 2: K-chunked exp + WMMA accumulate --------
  for (int kc = 0; kc < Nn / CK; ++kc) {
    int k0 = kc * CK;
    {   // build bf16 exp chunk in LDS (energy chunk is L2-resident now)
      const float* er = ebase + (size_t)(m0 + arow) * Nn + k0 + asub * 32;
      float mx = rowmax[arow];
      #pragma unroll
      for (int j = 0; j < 8; ++j) {
        float4 v = *(const float4*)(er + j * 4);
        bf16* dst = &attn[arow][asub * 32 + j * 4];
        dst[0] = (bf16)__expf(v.x - mx);
        dst[1] = (bf16)__expf(v.y - mx);
        dst[2] = (bf16)__expf(v.z - mx);
        dst[3] = (bf16)__expf(v.w - mx);
      }
    }
    __syncthreads();

    #pragma unroll 2
    for (int ks = 0; ks < CK / 32; ++ks) {
      v16bf bt = *(const v16bf*)(vbase + k0 + ks * 32);   // B: value_t row ct*16+lm
      #pragma unroll
      for (int rt = 0; rt < 4; ++rt) {
        const bf16* ap = &attn[rt * 16 + lm][ks * 32 + hi * 16];
        v8bf lo  = *(const v8bf*)(ap);        // halves[0..7]
        v8bf hic = *(const v8bf*)(ap + 8);    // halves[8..15] = K+8
        v16bf at = join16(lo, hic);
        acc[rt] = __builtin_amdgcn_wmma_f32_16x16x32_bf16(false, at, false, bt,
                                                          (short)0, acc[rt],
                                                          false, false);
      }
    }
    __syncthreads();   // protect attn[] before next chunk overwrites it
  }

  // -------- Epilogue: softmax 1/sum, gamma*out + 2x --------
  float g = gamma[0];
  int c = ct * 16 + lm;
  #pragma unroll
  for (int rt = 0; rt < 4; ++rt) {
    #pragma unroll
    for (int r = 0; r < 8; ++r) {
      int ml = rt * 16 + r + hi * 8;
      int mg = m0 + ml;
      size_t oidx = ((size_t)(b * Cc + c)) * Nn + mg;
      out[oidx] = g * acc[rt][r] * rowrcp[ml] + 2.0f * x[oidx];
    }
  }
}

// ---------------------------------------------------------------------------
extern "C" void kernel_launch(void* const* d_in, const int* in_sizes, int n_in,
                              void* d_out, int out_size, void* d_ws, size_t ws_size,
                              hipStream_t stream) {
  const float* energy = (const float*)d_in[0];   // [B, N, N]
  const float* x      = (const float*)d_in[1];   // [B, C, H, W]
  const float* W      = (const float*)d_in[2];   // [C, C]
  const float* bias   = (const float*)d_in[3];   // [C]
  const float* gamma  = (const float*)d_in[4];   // [1]
  float* out = (float*)d_out;

  bf16* xt = (bf16*)d_ws;                                        // [B][N][C] 8 MB
  bf16* vt = (bf16*)((char*)d_ws + (size_t)Bb * Nn * Cc * 2);    // [B][C][N] 8 MB

  xpose_kernel<<<dim3(Nn / 16, Cc / 16, Bb), dim3(16, 16), 0, stream>>>(x, xt);
  value_kernel<<<dim3(Nn / 16, Bb), 256, 0, stream>>>(xt, W, bias, vt);
  fused_kernel<<<dim3(Nn / MT, Bb), 512, 0, stream>>>(energy, vt, x, gamma, out);
}
